// SelfAttention_42073499631658
// MI455X (gfx1250) — matrix operand order repaired and checked by reading, not compiled, not run on hardware
//
#include <hip/hip_runtime.h>
#include <hip/hip_bf16.h>

// ---------------------------------------------------------------------------
// Self-attention, fused for MI455X (gfx1250), wave32 + v_wmma_f32_16x16x32_bf16
//   N=2, S=2048, E=1024, H=16, D=64
// Pipeline:
//   gemm_bias<f32->bf16> x3 : Q/K/V projections into workspace (bf16)
//   attn_fused             : scores -> head-axis softmax -> att@V (bf16 WMMA)
//                            K/Q tiles staged with GLOBAL_LOAD_ASYNC_TO_LDS
//   gemm_bias<bf16->f32>   : output projection into d_out
// Workspace layout (needs 32 MB):
//   [0,8M)   Q bf16   [8M,16M) K bf16   [16M,24M) V bf16   [24M,32M) attnOut bf16
// ---------------------------------------------------------------------------

#define NB   2
#define SS_  2048
#define EE_  1024
#define HH_  16
#define DD_  64

typedef __attribute__((ext_vector_type(16))) __bf16          v16bf;
typedef __attribute__((ext_vector_type(8)))  float           v8f;
typedef __attribute__((ext_vector_type(8)))  unsigned short  us8;

static __device__ __forceinline__ unsigned short f2bf(float f) {
    unsigned u = __builtin_bit_cast(unsigned, f);
    unsigned r = (u + 0x7FFFu + ((u >> 16) & 1u)) >> 16;   // round-to-nearest-even
    return (unsigned short)r;
}

static __device__ __forceinline__ v8f vzero8() {
    v8f v;
#pragma unroll
    for (int i = 0; i < 8; ++i) v[i] = 0.0f;
    return v;
}

// Async DMA: 16 bytes global -> LDS, tracked with ASYNCcnt (CDNA5 §15.18 op 98).
// ldsoff = LDS byte address (low 32 bits of generic shared address).
static __device__ __forceinline__ void async_g2l_b128(void* lds_ptr, const void* gptr) {
    unsigned           ldsoff = (unsigned)(size_t)lds_ptr;
    unsigned long long ga     = (unsigned long long)(size_t)gptr;
    asm volatile("global_load_async_to_lds_b128 %0, %1, off"
                 :: "v"(ldsoff), "v"(ga) : "memory");
}

static __device__ __forceinline__ void wait_asynccnt0() {
#if __has_builtin(__builtin_amdgcn_s_wait_asynccnt)
    __builtin_amdgcn_s_wait_asynccnt(0);
#else
    asm volatile("s_wait_asynccnt 0x0" ::: "memory");
#endif
}

// A-fragment (16x32 bf16, M = lane%16): element j -> K = kb0+j (contiguous).
// Caller passes p = rowStart + kb0 where kb0 = (lane>>4)*16. Two ds_load_b128.
static __device__ __forceinline__ v16bf ld_frag_gap(const unsigned short* p) {
    union { us8 u[2]; v16bf v; } x;
    x.u[0] = *(const us8*)(p);
    x.u[1] = *(const us8*)(p + 8);
    return x.v;
}

// B-fragment (32x16 bf16, N = lane%16): element j -> K = kbase+j (contiguous).
// Caller passes p = colRowStart + kbase where kbase = (lane>>4)*16. Two ds_load_b128.
static __device__ __forceinline__ v16bf ld_frag_seq(const unsigned short* p) {
    union { us8 u[2]; v16bf v; } x;
    x.u[0] = *(const us8*)(p);
    x.u[1] = *(const us8*)(p + 8);
    return x.v;
}

#define WMMA_BF16(a, b, c) \
    __builtin_amdgcn_wmma_f32_16x16x32_bf16(false, (a), false, (b), (short)0, (c), false, false)

// ---------------------------------------------------------------------------
// GEMM: C[M,NN] = cast(A[M,K]) * cast(B[K,NN]) + bias[NN]
// Block: 256 threads = 8 waves; block tile 128x64; wave tile 16x64; K-step 32.
// ---------------------------------------------------------------------------
template <bool A_IS_BF16, bool OUT_IS_BF16>
__global__ __launch_bounds__(256)
void gemm_bias_kernel(const void* __restrict__ Ap, const float* __restrict__ Bp,
                      const float* __restrict__ bias, void* __restrict__ Cp,
                      int M, int K, int NN) {
    __shared__ __align__(16) unsigned short As[128 * 32];   // 8 KB
    __shared__ __align__(16) unsigned short Bt[64 * 32];    // 4 KB (transposed: [n][k])

    const int tid  = threadIdx.x;
    const int lane = tid & 31;
    const int wv   = tid >> 5;
    const int m0   = blockIdx.x * 128;
    const int n0   = blockIdx.y * 64;

    v8f acc[4];
#pragma unroll
    for (int s = 0; s < 4; ++s) acc[s] = vzero8();

    const int nk = K >> 5;
    for (int kt = 0; kt < nk; ++kt) {
        // stage A tile (128x32) as bf16
#pragma unroll
        for (int j = 0; j < 16; ++j) {
            int idx = j * 256 + tid;
            int r = idx >> 5, c = idx & 31;
            size_t g = (size_t)(m0 + r) * K + kt * 32 + c;
            if (A_IS_BF16) As[r * 32 + c] = ((const unsigned short*)Ap)[g];
            else           As[r * 32 + c] = f2bf(((const float*)Ap)[g]);
        }
        // stage B tile transposed (Bt[n][k]), coalesced along n
#pragma unroll
        for (int j = 0; j < 8; ++j) {
            int idx = j * 256 + tid;
            int k = idx >> 6, n = idx & 63;
            Bt[n * 32 + k] = f2bf(Bp[(size_t)(kt * 32 + k) * NN + n0 + n]);
        }
        __syncthreads();

        v16bf a = ld_frag_gap(As + (wv * 16 + (lane & 15)) * 32 + ((lane >> 4) * 16));
#pragma unroll
        for (int s = 0; s < 4; ++s) {
            v16bf b = ld_frag_seq(Bt + (s * 16 + (lane & 15)) * 32 + ((lane >> 4) * 16));
            acc[s] = WMMA_BF16(a, b, acc[s]);
        }
        __syncthreads();
    }

    // epilogue: C/D layout -> element i: row M = i + 8*(lane>>4), col N = lane&15
#pragma unroll
    for (int s = 0; s < 4; ++s) {
#pragma unroll
        for (int i = 0; i < 8; ++i) {
            int row = m0 + wv * 16 + i + ((lane >> 4) * 8);
            int col = n0 + s * 16 + (lane & 15);
            float v = acc[s][i] + bias[col];
            if (OUT_IS_BF16) ((unsigned short*)Cp)[(size_t)row * NN + col] = f2bf(v);
            else             ((float*)Cp)[(size_t)row * NN + col] = v;
        }
    }
}

// ---------------------------------------------------------------------------
// Fused attention. 512 threads = 16 waves; wave h owns head h.
// Per block: one (n, q0..q0+15) tile; streams 128 k-tiles of 16.
// Softmax is across the 16 HEADS per (q,k) (reference quirk), scale 1/2048.
// Dynamic LDS: Qs 32K | Ks 32K | Vt 64K | Sc 16K | At 16K  = 160 KB
// ---------------------------------------------------------------------------
__global__ __launch_bounds__(512)
void attn_fused_kernel(const unsigned short* __restrict__ Qg,
                       const unsigned short* __restrict__ Kg,
                       const unsigned short* __restrict__ Vg,
                       const int* __restrict__ mask,
                       unsigned short* __restrict__ Og) {
    extern __shared__ __align__(16) unsigned char smem[];
    unsigned short* Qs = (unsigned short*)(smem);                    // [16][1024]
    unsigned short* Ks = (unsigned short*)(smem + 32768);            // [16][1024]
    unsigned short* Vt = (unsigned short*)(smem + 65536);            // [1024][32] (k-padded)
    float*          Sc = (float*)(smem + 131072);                    // [16][16][16]
    unsigned short* At = (unsigned short*)(smem + 147456);           // [16][16][32] (k-padded)

    const int tid  = threadIdx.x;
    const int lane = tid & 31;
    const int hh   = tid >> 5;                 // wave id == head id
    const int n    = blockIdx.x >> 7;
    const int q0   = (blockIdx.x & 127) << 4;

    // zero the K-pad halves once (never rewritten inside the loop)
    for (int i = tid; i < 1024 * 16; i += 512) {        // Vt[e][16..31] = 0
        int e = i >> 4, k = 16 + (i & 15);
        Vt[e * 32 + k] = 0;
    }
    for (int i = tid; i < 16 * 16 * 16; i += 512) {     // At[h][q][16..31] = 0
        int h = i >> 8, q = (i >> 4) & 15, k = 16 + (i & 15);
        At[h * 512 + q * 32 + k] = 0;
    }

    // Q tile (16 x 1024 bf16): async DMA straight into LDS (ASYNCcnt path)
    const unsigned short* Qb = Qg + ((size_t)n * SS_ + q0) * EE_;
    for (int i = tid; i < 2048; i += 512) {
        int r = i >> 7, c8 = i & 127;
        async_g2l_b128(Qs + r * 1024 + c8 * 8, Qb + r * 1024 + c8 * 8);
    }
    wait_asynccnt0();
    __syncthreads();

    v8f acc[4];
#pragma unroll
    for (int s = 0; s < 4; ++s) acc[s] = vzero8();

    for (int k0 = 0; k0 < SS_; k0 += 16) {
        const unsigned short* Kb = Kg + ((size_t)n * SS_ + k0) * EE_;
        const unsigned short* Vb = Vg + ((size_t)n * SS_ + k0) * EE_;

        // prefetch next k-tile (global_prefetch_b8; whole tile covered by 512 lanes)
        if (k0 + 16 < SS_) {
            __builtin_prefetch(Kb + 16 * EE_ + tid * 32, 0, 1);
            __builtin_prefetch(Vb + 16 * EE_ + tid * 32, 0, 1);
        }

        // K tile row-major: async global->LDS DMA (no VGPR round-trip)
        for (int i = tid; i < 2048; i += 512) {
            int r = i >> 7, c8 = i & 127;
            async_g2l_b128(Ks + r * 1024 + c8 * 8, Kb + r * 1024 + c8 * 8);
        }
        // V tile transposed into Vt[e][k] (transpose requires register path)
        for (int i = tid; i < 2048; i += 512) {
            int r = i >> 7, c8 = i & 127;
            us8 d = *(const us8*)(Vb + r * 1024 + c8 * 8);
#pragma unroll
            for (int j = 0; j < 8; ++j) Vt[(c8 * 8 + j) * 32 + r] = d[j];
        }
        wait_asynccnt0();
        __syncthreads();

        // ---- scores: head hh, 16x16 = Q_h(16x64) . K_h(16x64)^T, two K=32 steps
        v8f s = vzero8();
#pragma unroll
        for (int kk = 0; kk < 64; kk += 32) {
            int eb = hh * 64 + kk;
            v16bf a = ld_frag_gap(Qs + (lane & 15) * 1024 + eb + ((lane >> 4) * 16));
            v16bf b = ld_frag_seq(Ks + (lane & 15) * 1024 + eb + ((lane >> 4) * 16));
            s = WMMA_BF16(a, b, s);
        }
#pragma unroll
        for (int i = 0; i < 8; ++i) {
            int Mr = i + ((lane >> 4) * 8), Nc = lane & 15;
            Sc[hh * 256 + Mr * 16 + Nc] = s[i];
        }
        __syncthreads();

        // ---- softmax across the 16 heads for each (q,k); 256 lanes, one entry each
        if (tid < 256) {
            int q = tid >> 4, k = tid & 15;
            int mg = mask[(size_t)n * SS_ * SS_ + (size_t)(q0 + q) * SS_ + (k0 + k)];
            float x[16], mx = -3.4e38f;
#pragma unroll
            for (int h = 0; h < 16; ++h) {
                float v = Sc[h * 256 + q * 16 + k];
                v = (mg == 0) ? -1.0e20f : v;
                v *= (1.0f / (float)SS_);          // reference scales by seq length
                x[h] = v;
                mx = fmaxf(mx, v);
            }
            float sum = 0.0f, p[16];
#pragma unroll
            for (int h = 0; h < 16; ++h) { p[h] = __expf(x[h] - mx); sum += p[h]; }
            float inv = 1.0f / sum;
#pragma unroll
            for (int h = 0; h < 16; ++h) At[h * 512 + q * 32 + k] = f2bf(p[h] * inv);
        }
        __syncthreads();

        // ---- out_h(16x64) += att_h(16x32, K-padded) . V_h(32x16 per d-slice)
        v16bf a = ld_frag_gap(At + hh * 512 + (lane & 15) * 32 + ((lane >> 4) * 16));
#pragma unroll
        for (int dsl = 0; dsl < 4; ++dsl) {
            v16bf b = ld_frag_seq(Vt + (hh * 64 + dsl * 16 + (lane & 15)) * 32 +
                                  ((lane >> 4) * 16));
            acc[dsl] = WMMA_BF16(a, b, acc[dsl]);
        }
        __syncthreads();   // protect Ks/Vt/At before next iteration's overwrite
    }

    // write attn output (pre-Wo) as bf16: col = h*64 + d
    unsigned short* Ob = Og + ((size_t)n * SS_ + q0) * EE_;
#pragma unroll
    for (int dsl = 0; dsl < 4; ++dsl) {
#pragma unroll
        for (int i = 0; i < 8; ++i) {
            int row = i + ((lane >> 4) * 8);
            int col = hh * 64 + dsl * 16 + (lane & 15);
            Ob[(size_t)row * EE_ + col] = f2bf(acc[dsl][i]);
        }
    }
}

// ---------------------------------------------------------------------------
extern "C" void kernel_launch(void* const* d_in, const int* in_sizes, int n_in,
                              void* d_out, int out_size, void* d_ws, size_t ws_size,
                              hipStream_t stream) {
    const float* queries = (const float*)d_in[0];
    const float* keys    = (const float*)d_in[1];
    const float* values  = (const float*)d_in[2];
    const int*   mask    = (const int*)d_in[3];
    const float* Wq = (const float*)d_in[4];   const float* bq = (const float*)d_in[5];
    const float* Wk = (const float*)d_in[6];   const float* bk = (const float*)d_in[7];
    const float* Wv = (const float*)d_in[8];   const float* bv = (const float*)d_in[9];
    const float* Wo = (const float*)d_in[10];  const float* bo = (const float*)d_in[11];
    float* out = (float*)d_out;

    const size_t tok = (size_t)NB * SS_ * EE_;          // 4M elements
    unsigned short* Qw = (unsigned short*)d_ws;
    unsigned short* Kw = Qw + tok;
    unsigned short* Vw = Kw + tok;
    unsigned short* Ow = Vw + tok;                      // total 32 MB of ws

    const int Mrows = NB * SS_;                         // 4096
    dim3 ggrid(Mrows / 128, EE_ / 64);                  // 32 x 16
    dim3 gblk(256);

    // Q/K/V projections (f32 in, bf16 out)
    gemm_bias_kernel<false, true><<<ggrid, gblk, 0, stream>>>(queries, Wq, bq, Qw, Mrows, EE_, EE_);
    gemm_bias_kernel<false, true><<<ggrid, gblk, 0, stream>>>(keys,    Wk, bk, Kw, Mrows, EE_, EE_);
    gemm_bias_kernel<false, true><<<ggrid, gblk, 0, stream>>>(values,  Wv, bv, Vw, Mrows, EE_, EE_);

    // fused attention: 256 blocks (N * S/16), 512 threads (16 waves = 16 heads), 160KB LDS
    attn_fused_kernel<<<dim3(NB * (SS_ / 16)), dim3(512), 163840, stream>>>(Qw, Kw, Vw, mask, Ow);

    // output projection (bf16 in, f32 out)
    gemm_bias_kernel<true, false><<<ggrid, gblk, 0, stream>>>(Ow, Wo, bo, out, Mrows, EE_, EE_);
}
